// TemporalGCN_39728447488045
// MI455X (gfx1250) — matrix operand order repaired and checked by reading, not compiled, not run on hardware
//
#include <hip/hip_runtime.h>
#include <hip/hip_bf16.h>

#define NN      65536
#define BGRAPH  64
#define NMAX    1024
#define EDGES   2097152
#define DH      256
#define DOUT    128
#define BSTRIDE 40            // ushorts per LDS B-column: 32 data + 8 pad (80B, 16B-aligned)

// ---------------------------------------------------------------- bf16 utils
typedef __attribute__((ext_vector_type(16))) __bf16 bf16x16;
typedef __attribute__((ext_vector_type(8)))  float  f32x8;

union FragU { bf16x16 v; uint4 q[2]; };

__device__ __forceinline__ unsigned short f2bf(float f) {
    unsigned int u = __float_as_uint(f);
    unsigned int r = u + 0x7FFFu + ((u >> 16) & 1u);   // round-to-nearest-even
    return (unsigned short)(r >> 16);
}
__device__ __forceinline__ float bf2f_lo(unsigned int u) { return __uint_as_float(u << 16); }
__device__ __forceinline__ float bf2f_hi(unsigned int u) { return __uint_as_float(u & 0xffff0000u); }

// ---------------------------------------------------------------- degrees
__global__ void k_deg_init(float* __restrict__ deg) {
    deg[blockIdx.x * 256 + threadIdx.x] = 1.0f;        // self-loop
}

__global__ void k_deg_accum(const int* __restrict__ dst, float* __restrict__ deg, int E) {
    int e = blockIdx.x * 256 + threadIdx.x;
    if (e < E) unsafeAtomicAdd(&deg[dst[e]], 1.0f);
}

__global__ void k_dis(const float* __restrict__ deg, float* __restrict__ dis) {
    int i = blockIdx.x * 256 + threadIdx.x;
    dis[i] = rsqrtf(deg[i]);
}

// ---------------------------------------------------------------- bf16 staging
__global__ void k_cvt_x(const float* __restrict__ x, unsigned short* __restrict__ xbf) {
    size_t i = (size_t)blockIdx.x * 256 + threadIdx.x;
    xbf[i] = f2bf(x[i]);
}

// Wt[n][k] = bf16(W[k][n]) : col-major of W so WMMA B-fragments are contiguous
__global__ void k_prep_w(const float* __restrict__ W1, const float* __restrict__ W2,
                         unsigned short* __restrict__ W1t, unsigned short* __restrict__ W2t) {
    int n = blockIdx.x & 255;
    int which = blockIdx.x >> 8;
    int k = threadIdx.x;
    const float* W = which ? W2 : W1;
    unsigned short* Wt = which ? W2t : W1t;
    Wt[n * DH + k] = f2bf(W[k * DH + n]);
}

// ---------------------------------------------------------------- WMMA GEMM
// C_bf16[M x 256] = A_bf16[M x 256] @ W   (W given as Bt col-major: Bt[n*256+k])
// block = 256 threads = 8 waves laid out as 2 row-tiles x 4 col-strips.
// Each K-slice of B (256 cols x 32 k = 16KB) is staged once in LDS and
// consumed by all 8 waves via ds_load_b128.
__global__ __launch_bounds__(256) void k_gemm_bf16(
    const unsigned short* __restrict__ A,    // [M x 256] bf16 row-major
    const unsigned short* __restrict__ Bt,   // [256 x 256] bf16, Bt[n][k] = W[k][n]
    unsigned short* __restrict__ Cbf)        // [M x 256] bf16
{
    __shared__ unsigned short Bs[256 * BSTRIDE];       // 20 KB

    const int tid  = threadIdx.x;
    const int lane = tid & 31;
    const int wave = tid >> 5;
    const int lid  = lane & 15;
    const int half = lane >> 4;
    const int rt   = wave >> 2;                        // row tile 0..1
    const int cs   = (wave & 3) * 64;                  // col-strip base
    const int rowbase = blockIdx.x * 32 + rt * 16;

    f32x8 acc0 = {}, acc1 = {}, acc2 = {}, acc3 = {};
    const unsigned short* arow = A + (size_t)(rowbase + lid) * DH;

    for (int k = 0; k < DH; k += 32) {
        // cooperative stage: thread t copies column t's 32 bf16 of this K-slice
        {
            const unsigned short* gp = Bt + (size_t)tid * DH + k;
            uint4* lp = (uint4*)(Bs + tid * BSTRIDE);
            lp[0] = *(const uint4*)(gp);
            lp[1] = *(const uint4*)(gp + 8);
            lp[2] = *(const uint4*)(gp + 16);
            lp[3] = *(const uint4*)(gp + 24);
        }
        if (k + 32 < DH) __builtin_prefetch(arow + k + 32, 0, 1);
        __syncthreads();

        // A 16x32 fragment (lanes<16: K=0..15; lanes>=16: K=16..31)
        FragU a;
        a.q[0] = *(const uint4*)(arow + k + half * 16);
        a.q[1] = *(const uint4*)(arow + k + half * 16 + 8);

        // B 32x16 fragments from LDS (lanes<16 hold K=0..15, lanes>=16 K=16..31)
        FragU b0, b1, b2, b3;
        const unsigned short* bl = Bs + (cs + lid) * BSTRIDE + half * 16;
        b0.q[0] = *(const uint4*)(bl);                 b0.q[1] = *(const uint4*)(bl + 8);
        b1.q[0] = *(const uint4*)(bl + 16 * BSTRIDE);  b1.q[1] = *(const uint4*)(bl + 16 * BSTRIDE + 8);
        b2.q[0] = *(const uint4*)(bl + 32 * BSTRIDE);  b2.q[1] = *(const uint4*)(bl + 32 * BSTRIDE + 8);
        b3.q[0] = *(const uint4*)(bl + 48 * BSTRIDE);  b3.q[1] = *(const uint4*)(bl + 48 * BSTRIDE + 8);

        acc0 = __builtin_amdgcn_wmma_f32_16x16x32_bf16(false, a.v, false, b0.v, (short)0, acc0, false, false);
        acc1 = __builtin_amdgcn_wmma_f32_16x16x32_bf16(false, a.v, false, b1.v, (short)0, acc1, false, false);
        acc2 = __builtin_amdgcn_wmma_f32_16x16x32_bf16(false, a.v, false, b2.v, (short)0, acc2, false, false);
        acc3 = __builtin_amdgcn_wmma_f32_16x16x32_bf16(false, a.v, false, b3.v, (short)0, acc3, false, false);

        __syncthreads();
    }

    // C/D layout: lane VGPR r -> row = 8*half + r, col = lid; store bf16
    unsigned short* crow = Cbf + (size_t)(rowbase + half * 8) * DH;
#pragma unroll
    for (int r = 0; r < 8; ++r) {
        size_t ro = (size_t)r * DH;
        crow[ro + cs +  0 + lid] = f2bf(acc0[r]);
        crow[ro + cs + 16 + lid] = f2bf(acc1[r]);
        crow[ro + cs + 32 + lid] = f2bf(acc2[r]);
        crow[ro + cs + 48 + lid] = f2bf(acc3[r]);
    }
}

// ---------------------------------------------------------------- aggregation
__global__ void k_selfloop(const unsigned short* __restrict__ hwbf, const float* __restrict__ dis,
                           float* __restrict__ agg) {
    int n = blockIdx.x, d = threadIdx.x;
    float ds = dis[n];
    float v = bf2f_lo((unsigned int)hwbf[(size_t)n * DH + d]);
    agg[(size_t)n * DH + d] = v * ds * ds;
}

// one wave per edge, 8 dims per lane; gather bf16 (halved L2 traffic), f32 atomics
__global__ __launch_bounds__(256) void k_scatter(
    const int* __restrict__ src, const int* __restrict__ dst,
    const float* __restrict__ dis, const unsigned short* __restrict__ hwbf,
    float* __restrict__ agg, int E)
{
    int e = blockIdx.x * 8 + (threadIdx.x >> 5);
    if (e >= E) return;
    int lane = threadIdx.x & 31;
    int s = src[e], d = dst[e];
    float nrm = dis[s] * dis[d];
    uint4 q = *(const uint4*)(hwbf + (size_t)s * DH + lane * 8);
    float* ap = agg + (size_t)d * DH + lane * 8;
    unsafeAtomicAdd(ap + 0, bf2f_lo(q.x) * nrm);
    unsafeAtomicAdd(ap + 1, bf2f_hi(q.x) * nrm);
    unsafeAtomicAdd(ap + 2, bf2f_lo(q.y) * nrm);
    unsafeAtomicAdd(ap + 3, bf2f_hi(q.y) * nrm);
    unsafeAtomicAdd(ap + 4, bf2f_lo(q.z) * nrm);
    unsafeAtomicAdd(ap + 5, bf2f_hi(q.z) * nrm);
    unsafeAtomicAdd(ap + 6, bf2f_lo(q.w) * nrm);
    unsafeAtomicAdd(ap + 7, bf2f_hi(q.w) * nrm);
}

__global__ void k_bias_relu_bf16(const float* __restrict__ agg, const float* __restrict__ bias,
                                 unsigned short* __restrict__ out) {
    int d = threadIdx.x;
    size_t i = (size_t)blockIdx.x * DH + d;
    float v = agg[i] + bias[d];
    out[i] = f2bf(v > 0.f ? v : 0.f);
}

__global__ void k_bias_relu_f32(float* __restrict__ agg, const float* __restrict__ bias) {
    int d = threadIdx.x;
    size_t i = (size_t)blockIdx.x * DH + d;
    float v = agg[i] + bias[d];
    agg[i] = v > 0.f ? v : 0.f;
}

// ---------------------------------------------------------------- temporal pool (closed form) + head
__global__ void k_pool_sum(const float* __restrict__ h2, float* __restrict__ S) {
    int b = blockIdx.x, i = threadIdx.x;
    const float* p = h2 + (size_t)b * NMAX * DH + i;
    float acc = 0.f;
    for (int t = 0; t < NMAX; ++t) acc += p[(size_t)t * DH];
    S[b * DH + i] = acc;
}

// pooled[b,o] = (1/T) * sum_i [(w0+w1+w2)*S - w0*last - w2*first] + tb[o]
__global__ void k_pool_conv(const float* __restrict__ S, const float* __restrict__ h2,
                            const float* __restrict__ tw, const float* __restrict__ tb,
                            float* __restrict__ pooled) {
    int b = blockIdx.x, o = threadIdx.x;
    const float* first = h2 + (size_t)b * NMAX * DH;
    const float* last  = first + (size_t)(NMAX - 1) * DH;
    const float* Sb = S + b * DH;
    float acc = 0.f;
    for (int i = 0; i < DH; ++i) {
        const float* w = tw + ((size_t)o * DH + i) * 3;
        float w0 = w[0], w1 = w[1], w2 = w[2];
        acc += (w0 + w1 + w2) * Sb[i] - w0 * last[i] - w2 * first[i];
    }
    pooled[b * DH + o] = acc * (1.0f / NMAX) + tb[o];
}

__global__ void k_fc(const float* __restrict__ pooled, const float* __restrict__ fcw,
                     const float* __restrict__ fcb, float* __restrict__ out) {
    int b = blockIdx.x, j = threadIdx.x;
    const float* pb = pooled + b * DH;
    float acc = fcb[j];
    for (int o = 0; o < DH; ++o) acc += pb[o] * fcw[o * DOUT + j];
    out[b * DOUT + j] = acc;
}

// ---------------------------------------------------------------- launcher
extern "C" void kernel_launch(void* const* d_in, const int* in_sizes, int n_in,
                              void* d_out, int out_size, void* d_ws, size_t ws_size,
                              hipStream_t stream) {
    const float* x   = (const float*)d_in[0];
    const int*   ei  = (const int*)d_in[1];        // [2, E]: src then dst
    const float* W1  = (const float*)d_in[3];
    const float* b1  = (const float*)d_in[4];
    const float* W2  = (const float*)d_in[5];
    const float* b2  = (const float*)d_in[6];
    const float* tw  = (const float*)d_in[7];
    const float* tb  = (const float*)d_in[8];
    const float* fcw = (const float*)d_in[9];
    const float* fcb = (const float*)d_in[10];
    const int* srcI = ei;
    const int* dstI = ei + EDGES;
    float* out = (float*)d_out;

    char* ws = (char*)d_ws;
    const size_t KB = 1024, MB = 1024 * 1024;
    float*          deg    = (float*)(ws);                       // 256 KB
    float*          dis    = (float*)(ws + 256 * KB);            // 256 KB
    unsigned short* W1t    = (unsigned short*)(ws + 512 * KB);   // 128 KB
    unsigned short* W2t    = (unsigned short*)(ws + 640 * KB);   // 128 KB
    float*          S      = (float*)(ws + 768 * KB);            // 64 KB
    float*          pooled = (float*)(ws + 832 * KB);            // 64 KB
    unsigned short* xbf    = (unsigned short*)(ws + 1 * MB);     // 32 MB (reused as h1 bf16)
    unsigned short* hwbf   = (unsigned short*)(ws + 33 * MB);    // 32 MB bf16 GEMM out
    float*          agg    = (float*)(ws + 65 * MB);             // 64 MB (becomes h2)

    // --- normalization terms
    k_deg_init <<<NN / 256, 256, 0, stream>>>(deg);
    k_deg_accum<<<EDGES / 256, 256, 0, stream>>>(dstI, deg, EDGES);
    k_dis      <<<NN / 256, 256, 0, stream>>>(deg, dis);

    // --- bf16 staging
    k_cvt_x <<<NN * DH / 256, 256, 0, stream>>>(x, xbf);
    k_prep_w<<<512, 256, 0, stream>>>(W1, W2, W1t, W2t);

    // --- GCN layer 1
    k_gemm_bf16<<<NN / 32, 256, 0, stream>>>(xbf, W1t, hwbf);
    k_selfloop <<<NN, DH, 0, stream>>>(hwbf, dis, agg);
    k_scatter  <<<EDGES / 8, 256, 0, stream>>>(srcI, dstI, dis, hwbf, agg, EDGES);
    k_bias_relu_bf16<<<NN, DH, 0, stream>>>(agg, b1, xbf);       // h1 (bf16) overwrites xbf

    // --- GCN layer 2
    k_gemm_bf16<<<NN / 32, 256, 0, stream>>>(xbf, W2t, hwbf);
    k_selfloop <<<NN, DH, 0, stream>>>(hwbf, dis, agg);
    k_scatter  <<<EDGES / 8, 256, 0, stream>>>(srcI, dstI, dis, hwbf, agg, EDGES);
    k_bias_relu_f32<<<NN, DH, 0, stream>>>(agg, b2);             // h2 in place

    // --- temporal conv + mean pool (closed form) + FC head
    k_pool_sum <<<BGRAPH, DH, 0, stream>>>(agg, S);
    k_pool_conv<<<BGRAPH, DH, 0, stream>>>(S, agg, tw, tb, pooled);
    k_fc       <<<BGRAPH, DOUT, 0, stream>>>(pooled, fcw, fcb, out);
}
